// TransformerEncoderLayer_9552007266481
// MI455X (gfx1250) — compile-verified, matrix operand order checked by reading, not run on hardware
//
#include <hip/hip_runtime.h>
#include <math.h>

// ---------------------------------------------------------------------------
// TransformerEncoderLayer for shogi boards on MI455X (gfx1250).
// All GEMMs use v_wmma_f32_16x16x32_f16 (f16 in, f32 accumulate).
// Per-board workgroup keeps qkv / scores / ff-hidden in LDS (WGP has 320KB).
// B-fragments that are invariant across M-tiles are hoisted into VGPRs so
// WMMAs issue back-to-back instead of waiting on per-tile LDS loads.
// d_out is the inter-kernel activation buffer (y -> z in place);
// d_ws only stores pre-packed f16 weight fragments + BN stats.
// ---------------------------------------------------------------------------

#define D_   256
#define NH_  8
#define DEP_ 32
#define FF_  1024
#define P_   81
#define PP_  96     // tokens padded to 6 tiles of 16
#define GQKV 96     // 3*DEP outputs per conv group

typedef __attribute__((ext_vector_type(16))) _Float16 v16h;
typedef __attribute__((ext_vector_type(8)))  float    v8f;

union HFrag { v16h v; _Float16 h[16]; unsigned u[8]; };
union FAcc  { v8f  v; float    f[8];                 };
union H8    { _Float16 h[8]; uint4 q;                };

// Build a 16x32 f16 fragment (A- or B-style) from an LDS array laid out
// [row][k] with k contiguous.  Mapping follows ISA 16-bit A-matrix table:
//   VGPR v, elem e: k = k0 + ((v&4)<<2) + ((v&3)<<1) + ((lane>=16)?8:0) + e
//   row = row0 + (lane & 15)
__device__ __forceinline__ v16h frag_ld_lds(const _Float16* arr, int row0,
                                            int ld, int k0, int lane) {
  const int row = row0 + (lane & 15);
  const int hi  = (lane >> 4) & 1;
  const _Float16* p = arr + row * ld + k0 + (hi << 3);
  HFrag f;
#pragma unroll
  for (int v = 0; v < 8; ++v) {
    const int off = ((v & 4) << 2) | ((v & 3) << 1);
    f.u[v] = *(const unsigned*)(p + off);
  }
  return f.v;
}

// Load a pre-packed A fragment from global memory: 32 lanes x 16 halves,
// lane-contiguous (32B per lane -> two global_load_b128).
__device__ __forceinline__ v16h frag_ld_packed(const _Float16* pw,
                                               int fragIdx, int lane) {
  const uint4* p = (const uint4*)(pw + ((long)fragIdx * 32 + lane) * 16);
  uint4 a = p[0], b = p[1];
  HFrag f;
  f.u[0] = a.x; f.u[1] = a.y; f.u[2] = a.z; f.u[3] = a.w;
  f.u[4] = b.x; f.u[5] = b.y; f.u[6] = b.z; f.u[7] = b.w;
  return f.v;
}

// ---------------------------------------------------------------------------
// Kernel 0: pack an f32 weight matrix W[M,K] (row-major) into f16 WMMA
// A-fragments.  frag index = mt*(K/32) + kt;  1KB per fragment.
// ---------------------------------------------------------------------------
__global__ void pack_kernel(const float* __restrict__ W,
                            _Float16* __restrict__ out, int M, int K) {
  const int nfk = K >> 5;
  const int total = (M >> 4) * nfk * 32;
  for (int i = blockIdx.x * blockDim.x + threadIdx.x; i < total;
       i += gridDim.x * blockDim.x) {
    const int lane = i & 31;
    const int frag = i >> 5;
    const int kt = frag % nfk, mt = frag / nfk;
    const int m  = mt * 16 + (lane & 15);
    const int hi = (lane >> 4) & 1;
    _Float16* o = out + (long)i * 16;
    const float* w = W + (long)m * K + kt * 32 + (hi << 3);
#pragma unroll
    for (int v = 0; v < 8; ++v) {
      const int off = ((v & 4) << 2) | ((v & 3) << 1);
      o[v * 2]     = (_Float16)w[off];
      o[v * 2 + 1] = (_Float16)w[off + 1];
    }
  }
}

// ---------------------------------------------------------------------------
// Kernel 1: fused attention block for one board per workgroup (8 waves).
// ---------------------------------------------------------------------------
__global__ void attn_kernel(const float* __restrict__ x,
                            const _Float16* __restrict__ qkvP,
                            const float* __restrict__ relw1,
                            const float* __restrict__ relw2,
                            const _Float16* __restrict__ owP,
                            float* __restrict__ y) {
  extern __shared__ char smem[];
  _Float16* xT   = (_Float16*)smem;                 // [96][256]  x (later att)
  _Float16* qh   = xT + PP_ * D_;                   // [8][96][32] q  [t][d]
  _Float16* kh   = qh + NH_ * PP_ * DEP_;           // [8][96][32] k+r
  _Float16* vh   = kh + NH_ * PP_ * DEP_;           // [8][32][96] v  [d][t]
  float*    sscr = (float*)(vh + NH_ * DEP_ * PP_); // [8][16][96] scores
  _Float16* awsc = (_Float16*)(sscr + NH_ * 16 * PP_); // [8][16][96] softmax
  float*    tpart= (float*)(awsc + NH_ * 16 * PP_); // [256] partials
  float*    tvec = tpart + 256;                     // [32] rel hidden

  const int tid  = threadIdx.x;
  const int wave = tid >> 5, lane = tid & 31;
  const int hi   = (lane >> 4) & 1;
  const long xb  = (long)blockIdx.x * (D_ * P_);

  // ---- stage x into LDS as f16, token-major (k=channel contiguous) ----
  for (int i = tid; i < PP_ * D_; i += 256) {
    const int t = i >> 8, c = i & 255;
    const float v = (t < P_) ? x[xb + c * P_ + t] : 0.0f;
    xT[i] = (_Float16)v;
  }
  __syncthreads();

  // ---- grouped 1x1 conv QKV: wave = group g ----
  {
    const int g = wave;
    v16h bfr[6];
#pragma unroll
    for (int nt = 0; nt < 6; ++nt)
      bfr[nt] = frag_ld_lds(xT, nt * 16, D_, g * DEP_, lane);
    for (int mt = 0; mt < 6; ++mt) {
      const v16h a = frag_ld_packed(qkvP, g * 6 + mt, lane);  // K=32: nfk==1
      // destination octet (8 consecutive output rows) is uniform over e:
      const int f0    = g * GQKV + mt * 16 + (hi << 3); // multiple of 8
      const int which = f0 >> 8;                        // 0=q 1=k 2=v
      const int rem   = f0 & 255;
      const int h2 = rem >> 5, d0 = rem & 31;           // d0 multiple of 8
#pragma unroll
      for (int nt = 0; nt < 6; ++nt) {
        v8f acc = {};
        acc = __builtin_amdgcn_wmma_f32_16x16x32_f16(false, a, false, bfr[nt],
                                                     (short)0, acc, false, false);
        FAcc r; r.v = acc;
        H8 hv;
#pragma unroll
        for (int e = 0; e < 8; ++e) hv.h[e] = (_Float16)r.f[e];
        const int t = nt * 16 + (lane & 15);
        if (which != 2) {
          // q/k: [t][d] layout, 8 contiguous halves, 16B aligned
          _Float16* dst = (which == 0 ? qh : kh) + (h2 * PP_ + t) * DEP_ + d0;
          *(uint4*)dst = hv.q;
        } else {
          // v: [d][t] transposed layout, strided scalar stores
          _Float16* dst = vh + (h2 * DEP_ + d0) * PP_ + t;
#pragma unroll
          for (int e = 0; e < 8; ++e) dst[e * PP_] = hv.h[e];
        }
      }
    }
  }

  // ---- rel hidden: tvec[32] = rel_w1 @ x_flat (cooperative dot) ----
  {
    const int j = tid & 31, chunk = tid >> 5;  // chunk = 32 channels
    const float* w1 = relw1 + (long)j * (D_ * P_);
    float acc = 0.f;
    const int c0 = chunk * 32;
    for (int c = c0; c < c0 + 32; ++c) {
      const float* w1c = w1 + c * P_;
      const _Float16* xc = xT + c;
      for (int t = 0; t < P_; ++t) acc += w1c[t] * (float)xc[t * D_];
    }
    tpart[tid] = acc;
  }
  __syncthreads();
  if (tid < 32) {
    float s = 0.f;
    for (int k2 = 0; k2 < 8; ++k2) s += tpart[k2 * 32 + tid];
    tvec[tid] = s;
  }
  __syncthreads();

  // ---- r = rel_w2 @ tvec; fold into K of every head (q@k + q@r = q@(k+r)) --
  for (int i = tid; i < DEP_ * P_; i += 256) {
    const int d = i / P_, t = i - d * P_;
    const float* w2 = relw2 + (long)i * 32;
    float rv = 0.f;
#pragma unroll
    for (int j = 0; j < 32; ++j) rv += w2[j] * tvec[j];
    for (int h2 = 0; h2 < NH_; ++h2) {
      const int idx = (h2 * PP_ + t) * DEP_ + d;
      kh[idx] = (_Float16)((float)kh[idx] + rv);
    }
  }
  __syncthreads();

  // ---- attention: wave = head ----
  {
    const int h = wave;
    const _Float16* qB = qh + h * PP_ * DEP_;
    const _Float16* kB = kh + h * PP_ * DEP_;
    const _Float16* vB = vh + h * DEP_ * PP_;
    float*    sw = sscr + wave * 16 * PP_;
    _Float16* aw = awsc + wave * 16 * PP_;
    const float scale = 0.17677669529663687f;  // 1/sqrt(32)

    // hoist K-tiles and V-tiles into registers (shared across all mt)
    v16h kfr[6];
#pragma unroll
    for (int nt = 0; nt < 6; ++nt)
      kfr[nt] = frag_ld_lds(kB, nt * 16, DEP_, 0, lane);
    v16h vfr[2][3];
#pragma unroll
    for (int dt = 0; dt < 2; ++dt)
#pragma unroll
      for (int ks = 0; ks < 3; ++ks)
        vfr[dt][ks] = frag_ld_lds(vB, dt * 16, PP_, ks * 32, lane);

    for (int mt = 0; mt < 6; ++mt) {
      const v16h aq = frag_ld_lds(qB, mt * 16, DEP_, 0, lane);
#pragma unroll
      for (int nt = 0; nt < 6; ++nt) {          // scores tile (K=32, 1 step)
        v8f acc = {};
        acc = __builtin_amdgcn_wmma_f32_16x16x32_f16(false, aq, false, kfr[nt],
                                                     (short)0, acc, false, false);
        FAcc r; r.v = acc;
        float* swp = sw + (hi << 3) * PP_ + nt * 16 + (lane & 15);
#pragma unroll
        for (int e = 0; e < 8; ++e) swp[e * PP_] = r.f[e] * scale;
      }
      // row softmax over 81 valid columns (lanes 0..15, one row each)
      if (lane < 16) {
        float mx = -1e30f;
        for (int j2 = 0; j2 < P_; ++j2) mx = fmaxf(mx, sw[lane * PP_ + j2]);
        float s = 0.f;
        for (int j2 = 0; j2 < P_; ++j2) {
          const float e2 = __expf(sw[lane * PP_ + j2] - mx);
          sw[lane * PP_ + j2] = e2; s += e2;
        }
        const float inv = 1.f / s;
        for (int j2 = 0; j2 < PP_; ++j2)
          aw[lane * PP_ + j2] =
              (j2 < P_) ? (_Float16)(sw[lane * PP_ + j2] * inv) : (_Float16)0.f;
      }
      // att tile = aw @ v   (att overwrites xT: [t][c] layout for o-proj B)
      v16h afr[3];
#pragma unroll
      for (int ks = 0; ks < 3; ++ks)
        afr[ks] = frag_ld_lds(aw, 0, PP_, ks * 32, lane);
#pragma unroll
      for (int dt = 0; dt < 2; ++dt) {
        v8f acc = {};
#pragma unroll
        for (int ks = 0; ks < 3; ++ks)
          acc = __builtin_amdgcn_wmma_f32_16x16x32_f16(false, afr[ks], false,
                                                       vfr[dt][ks], (short)0,
                                                       acc, false, false);
        FAcc r; r.v = acc;
        _Float16* dst = xT + (mt * 16 + (hi << 3)) * D_ + h * DEP_ + dt * 16
                        + (lane & 15);
#pragma unroll
        for (int e = 0; e < 8; ++e) dst[e * D_] = (_Float16)r.f[e];
      }
    }
  }
  __syncthreads();

  // ---- output projection + residual: wave owns 32 output channels ----
  for (int nt = 0; nt < 6; ++nt) {
    v16h bfr[8];
#pragma unroll
    for (int kt = 0; kt < 8; ++kt)
      bfr[kt] = frag_ld_lds(xT, nt * 16, D_, kt * 32, lane);
    const int t = nt * 16 + (lane & 15);
#pragma unroll
    for (int mi = 0; mi < 2; ++mi) {
      const int mt2 = wave * 2 + mi;
      v8f acc = {};
#pragma unroll
      for (int kt = 0; kt < 8; ++kt) {
        const v16h a = frag_ld_packed(owP, mt2 * 8 + kt, lane);
        acc = __builtin_amdgcn_wmma_f32_16x16x32_f16(false, a, false, bfr[kt],
                                                     (short)0, acc, false, false);
      }
      FAcc r; r.v = acc;
      if (t < P_) {
        const int c0 = mt2 * 16 + (hi << 3);
        float* dst = y + xb + c0 * P_ + t;
        const float* src = x + xb + c0 * P_ + t;
#pragma unroll
        for (int e = 0; e < 8; ++e) dst[e * P_] = r.f[e] + src[e * P_];
      }
    }
  }
}

// ---------------------------------------------------------------------------
// Kernel 2/4: per-channel BN stats -> st[c] = g*rsqrt(var+eps),
//             st[256+c] = b - mean*scale.   One block per channel.
// ---------------------------------------------------------------------------
__global__ void bn_stats_kernel(const float* __restrict__ z,
                                const float* __restrict__ g,
                                const float* __restrict__ bb,
                                float* __restrict__ st, int Bn) {
  __shared__ float s1[256], s2[256];
  const int c = blockIdx.x;
  float a = 0.f, q = 0.f;
  for (int bi = threadIdx.x; bi < Bn; bi += 256) {
    const float* p = z + (long)bi * (D_ * P_) + c * P_;
    for (int t = 0; t < P_; ++t) { const float v = p[t]; a += v; q += v * v; }
  }
  s1[threadIdx.x] = a; s2[threadIdx.x] = q;
  __syncthreads();
  for (int s = 128; s > 0; s >>= 1) {
    if (threadIdx.x < s) {
      s1[threadIdx.x] += s1[threadIdx.x + s];
      s2[threadIdx.x] += s2[threadIdx.x + s];
    }
    __syncthreads();
  }
  if (threadIdx.x == 0) {
    const float n = (float)Bn * (float)P_;
    const float mean = s1[0] / n;
    const float var  = s2[0] / n - mean * mean;
    const float sc   = g[c] * rsqrtf(var + 1e-5f);
    st[c]      = sc;
    st[D_ + c] = bb[c] - mean * sc;
  }
}

// ---------------------------------------------------------------------------
// Kernel 3: fused FF block, in place on d_out:
//   x1 = BN1(y) (f16 in LDS) -> hid = gelu(ff_w1 @ x1) (LDS) ->
//   z = ff_w2 @ hid + x1
// ---------------------------------------------------------------------------
__global__ void ff_kernel(float* __restrict__ z, const float* __restrict__ st,
                          const _Float16* __restrict__ f1P,
                          const _Float16* __restrict__ f2P) {
  extern __shared__ char smem[];
  _Float16* x1T = (_Float16*)smem;        // [96][256]
  _Float16* hid = x1T + PP_ * D_;         // [96][1024]
  const int tid  = threadIdx.x;
  const int wave = tid >> 5, lane = tid & 31;
  const int hi   = (lane >> 4) & 1;
  const long xb  = (long)blockIdx.x * (D_ * P_);

  for (int i = tid; i < PP_ * D_; i += 256) {
    const int t = i >> 8, c = i & 255;
    float v = 0.f;
    if (t < P_) v = z[xb + c * P_ + t] * st[c] + st[D_ + c];
    x1T[i] = (_Float16)v;
  }
  __syncthreads();

  // FF1 + exact GELU: nt-outer, B-fragments held in registers for all mt
  for (int nt = 0; nt < 6; ++nt) {
    v16h bfr[8];
#pragma unroll
    for (int kt = 0; kt < 8; ++kt)
      bfr[kt] = frag_ld_lds(x1T, nt * 16, D_, kt * 32, lane);
    const int t = nt * 16 + (lane & 15);
    for (int mt = wave; mt < 64; mt += 8) {
      __builtin_prefetch(f1P + ((long)((mt + 8 < 64 ? mt + 8 : mt) * 8) * 32
                                + lane) * 16, 0, 1);
      v8f acc = {};
#pragma unroll
      for (int kt = 0; kt < 8; ++kt) {
        const v16h a = frag_ld_packed(f1P, mt * 8 + kt, lane);
        acc = __builtin_amdgcn_wmma_f32_16x16x32_f16(false, a, false, bfr[kt],
                                                     (short)0, acc, false, false);
      }
      FAcc r; r.v = acc;
      H8 hv;
#pragma unroll
      for (int e = 0; e < 8; ++e) {
        const float u = r.f[e];
        hv.h[e] = (_Float16)(0.5f * u * (1.f + erff(u * 0.70710678118654752f)));
      }
      // 8 contiguous ff-channels at [t][f0], 16B aligned -> one ds_store_b128
      const int f0 = mt * 16 + (hi << 3);
      *(uint4*)(hid + t * FF_ + f0) = hv.q;
    }
  }
  __syncthreads();

  // FF2 + residual: K chunked by 8; B-fragments shared by both output tiles
  for (int nt = 0; nt < 6; ++nt) {
    const int t = nt * 16 + (lane & 15);
    v8f acc0 = {}, acc1 = {};
    for (int ktc = 0; ktc < 4; ++ktc) {
      v16h bfr[8];
#pragma unroll
      for (int k8 = 0; k8 < 8; ++k8)
        bfr[k8] = frag_ld_lds(hid, nt * 16, FF_, (ktc * 8 + k8) * 32, lane);
#pragma unroll
      for (int k8 = 0; k8 < 8; ++k8) {
        const int kt = ktc * 8 + k8;
        const v16h a0 = frag_ld_packed(f2P, (wave * 2) * 32 + kt, lane);
        acc0 = __builtin_amdgcn_wmma_f32_16x16x32_f16(false, a0, false, bfr[k8],
                                                      (short)0, acc0, false, false);
        const v16h a1 = frag_ld_packed(f2P, (wave * 2 + 1) * 32 + kt, lane);
        acc1 = __builtin_amdgcn_wmma_f32_16x16x32_f16(false, a1, false, bfr[k8],
                                                      (short)0, acc1, false, false);
      }
    }
    if (t < P_) {
#pragma unroll
      for (int mi = 0; mi < 2; ++mi) {
        FAcc r; r.v = (mi == 0) ? acc0 : acc1;
        const int c0 = (wave * 2 + mi) * 16 + (hi << 3);
        float* dst = z + xb + c0 * P_ + t;
        const _Float16* x1p = x1T + t * D_ + c0;
#pragma unroll
        for (int e = 0; e < 8; ++e) dst[e * P_] = r.f[e] + (float)x1p[e];
      }
    }
  }
}

// Kernel 5: final BN2 normalize, in place.  block = (b*256 + c).
__global__ void bn_apply_kernel(float* __restrict__ z,
                                const float* __restrict__ st) {
  const long bc = blockIdx.x;
  const int c = (int)(bc & 255);
  const float sc = st[c], bi = st[D_ + c];
  float* p = z + bc * P_;
  const int t = threadIdx.x;
  if (t < P_) p[t] = p[t] * sc + bi;
}

// ---------------------------------------------------------------------------
extern "C" void kernel_launch(void* const* d_in, const int* in_sizes, int n_in,
                              void* d_out, int out_size, void* d_ws,
                              size_t ws_size, hipStream_t stream) {
  const float* x      = (const float*)d_in[0];
  const float* qkv_w  = (const float*)d_in[1];
  const float* rel_w1 = (const float*)d_in[2];
  const float* rel_w2 = (const float*)d_in[3];
  const float* o_w    = (const float*)d_in[4];
  const float* ff_w1  = (const float*)d_in[5];
  const float* ff_w2  = (const float*)d_in[6];
  const float* bn1_g  = (const float*)d_in[7];
  const float* bn1_b  = (const float*)d_in[8];
  const float* bn2_g  = (const float*)d_in[9];
  const float* bn2_b  = (const float*)d_in[10];
  const int Bn = in_sizes[0] / (D_ * P_);
  float* out = (float*)d_out;

  // workspace: packed f16 weight fragments + BN stats (~1.2 MB total)
  char* w = (char*)d_ws;
  _Float16* qkvP = (_Float16*)w; w += (size_t)768 * 32 * 2;
  _Float16* owP  = (_Float16*)w; w += (size_t)256 * 256 * 2;
  _Float16* f1P  = (_Float16*)w; w += (size_t)1024 * 256 * 2;
  _Float16* f2P  = (_Float16*)w; w += (size_t)256 * 1024 * 2;
  float* bn1st   = (float*)w;    w += (size_t)512 * 4;
  float* bn2st   = (float*)w;

  pack_kernel<<<6,  256, 0, stream>>>(qkv_w, qkvP, 768, 32);
  pack_kernel<<<16, 256, 0, stream>>>(o_w,   owP,  256, 256);
  pack_kernel<<<64, 256, 0, stream>>>(ff_w1, f1P,  1024, 256);
  pack_kernel<<<64, 256, 0, stream>>>(ff_w2, f2P,  256, 1024);

  const size_t smem1 = (size_t)(PP_ * D_ * 2)             // xT / att
                     + (size_t)(3 * NH_ * PP_ * DEP_ * 2) // q,k,v
                     + (size_t)(NH_ * 16 * PP_ * 4)       // scores scratch
                     + (size_t)(NH_ * 16 * PP_ * 2)       // softmax f16
                     + (size_t)(256 * 4 + 32 * 4);        // rel partials
  attn_kernel<<<Bn, 256, smem1, stream>>>(x, qkvP, rel_w1, rel_w2, owP, out);
  bn_stats_kernel<<<D_, 256, 0, stream>>>(out, bn1_g, bn1_b, bn1st, Bn);

  const size_t smem2 = (size_t)(PP_ * D_ * 2) + (size_t)(PP_ * FF_ * 2);
  ff_kernel<<<Bn, 256, smem2, stream>>>(out, bn1st, f1P, f2P);
  bn_stats_kernel<<<D_, 256, 0, stream>>>(out, bn2_g, bn2_b, bn2st, Bn);

  bn_apply_kernel<<<(unsigned)Bn * 256, 96, 0, stream>>>(out, bn2st);
}
